// Block_63866163692094
// MI455X (gfx1250) — matrix operand order repaired and checked by reading, not compiled, not run on hardware
//
#include <hip/hip_runtime.h>

typedef __attribute__((ext_vector_type(16))) _Float16 v16h;
typedef __attribute__((ext_vector_type(8)))  _Float16 v8h;
typedef __attribute__((ext_vector_type(8)))  float    v8f;

#define DEVI __device__ __forceinline__

// problem dims (fixed by the reference)
#define Bq  4
#define Tq  1024
#define Mq  256
#define Cq  1024
#define Hq  16
#define FFq 4096
#define HDq 64

// GEMM epilogue / operand flags
#define F_BIAS   1
#define F_RES    2
#define F_OUTF16 4
#define F_BTRANS 8

constexpr int BM = 128, BN = 128, BK = 32;
constexpr int LSTR = 48; // padded LDS row stride in halves (96B -> 16B aligned)

DEVI v16h frag_cat(const _Float16* p0, const _Float16* p1) {
  v8h a = *(const v8h*)p0;
  v8h b = *(const v8h*)p1;
  v16h r;
#pragma unroll
  for (int e = 0; e < 8; ++e) { r[e] = a[e]; r[e + 8] = b[e]; }
  return r;
}

// LDS byte offset of a __shared__ object = low 32 bits of its flat address
DEVI unsigned lds_off_of(const void* p) {
  return (unsigned)(unsigned long long)p;
}

// 32B memory->LDS async copy (two b128 transfers; INST_OFFSET is added to
// both the LDS and the global address per the CDNA5 TDM/async spec)
DEVI void async_copy32(unsigned lds_b, const _Float16* gsrc) {
  asm volatile(
      "global_load_async_to_lds_b128 %0, %1, off\n\t"
      "global_load_async_to_lds_b128 %0, %1, off offset:16"
      :: "v"(lds_b), "v"(gsrc) : "memory");
}

DEVI void wait_async0() {
  asm volatile("s_wait_asynccnt 0" ::: "memory");
}

// ---------------------------------------------------------------------------
// Generic batched WMMA GEMM:  out = alpha * A(f16) * B(f16)^T? (+bias)(+res)
//   A: [M,K] row-major (lda), per-batch offset zo*aOuter + zi*aInner
//   B: if !BTRANS: source is [N,K] row-major (ldb)  (weights / K-matrix)
//      if  BTRANS: source is [K,N] row-major (ldb)  (V-matrix)
//   out: f32 or f16 per F_OUTF16; res is f32 with identical indexing.
// Non-BTRANS launches in this file always have M,N multiples of 128, K of 32,
// so the async staging path needs no edge masking (A-staging likewise).
// ---------------------------------------------------------------------------
template <int FLAGS>
__global__ __launch_bounds__(256) void gemm_wmma(
    const _Float16* __restrict__ A, long aOuter, long aInner, int lda,
    const _Float16* __restrict__ Bm, long bOuter, long bInner, int ldb,
    const float* __restrict__ bias, const float* __restrict__ res,
    void* __restrict__ out, long oOuter, long oInner, int ldo,
    int M, int N, int K, float alpha, int zdiv) {
  const int z  = blockIdx.z;
  const int zo = z / zdiv;
  const int zi = z - zo * zdiv;
  const _Float16* Ab = A  + zo * aOuter + zi * aInner;
  const _Float16* Bb = Bm + zo * bOuter + zi * bInner;
  const long oOff = zo * oOuter + zi * oInner;

  const int row0 = blockIdx.y * BM;
  const int col0 = blockIdx.x * BN;
  const int tid  = threadIdx.x;
  const int lane = tid & 31;
  const int wave = tid >> 5;
  const int waveM = wave >> 2;   // 0..1 -> 64 rows each
  const int waveN = wave & 3;    // 0..3 -> 32 cols each

  __shared__ _Float16 sA[2][BM * LSTR];
  __shared__ _Float16 sB[2][BN * LSTR];

  const v8f vzero = {0.f, 0.f, 0.f, 0.f, 0.f, 0.f, 0.f, 0.f};
  v8f acc[4][2];
#pragma unroll
  for (int mi = 0; mi < 4; ++mi)
#pragma unroll
    for (int ni = 0; ni < 2; ++ni) acc[mi][ni] = vzero;

  // staging coordinates: each thread owns 16 halves (32B) of the 128x32 tile
  const int sr = tid >> 1;           // 0..127 (row of A-tile / n of B-tile)
  const int sc = (tid & 1) * 16;     // 0 or 16 (k within tile)

  auto stage = [&](int k0, int buf) {
    // ---- A tile [BM x BK], K-contiguous -> async ----
    async_copy32(lds_off_of(&sA[buf][sr * LSTR + sc]),
                 Ab + (long)(row0 + sr) * lda + k0 + sc);
    if (!(FLAGS & F_BTRANS)) {
      // ---- B tile from [N,K] row-major -> async ----
      async_copy32(lds_off_of(&sB[buf][sr * LSTR + sc]),
                   Bb + (long)(col0 + sr) * ldb + k0 + sc);
    } else {
      // ---- B tile from [K,N] row-major: transpose into sB[n][k] ----
      const int kk = tid >> 3;           // 0..31
      const int nb = (tid & 7) * 16;     // 0..112
      const _Float16* src = Bb + (long)(k0 + kk) * ldb + col0 + nb;
#pragma unroll
      for (int j = 0; j < 16; ++j) {
        _Float16 v = (_Float16)0.f;
        if (col0 + nb + j < N) v = src[j];
        sB[buf][(nb + j) * LSTR + kk] = v;
      }
    }
  };

  // prologue: fill buffer 0
  stage(0, 0);
  wait_async0();
  __syncthreads();

  int buf = 0;
  for (int k0 = 0; k0 < K; k0 += BK) {
    // prefetch next tile into the other buffer while computing
    if (k0 + BK < K) stage(k0 + BK, buf ^ 1);

    // ---- fragments + 8 WMMAs per wave ----
    const int g  = lane >> 4;
    const int lr = lane & 15;
    v16h af[4];
#pragma unroll
    for (int mi = 0; mi < 4; ++mi) {
      const _Float16* p = &sA[buf][(waveM * 64 + mi * 16 + lr) * LSTR + 8 * g];
      af[mi] = frag_cat(p, p + 16);
    }
#pragma unroll
    for (int ni = 0; ni < 2; ++ni) {
      const _Float16* p = &sB[buf][(waveN * 32 + ni * 16 + lr) * LSTR + 8 * g];
      v16h bf = frag_cat(p, p + 16);
#pragma unroll
      for (int mi = 0; mi < 4; ++mi) {
        acc[mi][ni] = __builtin_amdgcn_wmma_f32_16x16x32_f16(
            false, af[mi], false, bf, (short)0, acc[mi][ni], false, false);
      }
    }

    wait_async0();      // next buffer fully resident in LDS
    __syncthreads();    // all waves done reading current buffer
    buf ^= 1;
  }

  // ---- epilogue: scale (+bias) (+residual), store f32 or f16 ----
  const int g  = lane >> 4;
  const int lr = lane & 15;
#pragma unroll
  for (int mi = 0; mi < 4; ++mi) {
#pragma unroll
    for (int ni = 0; ni < 2; ++ni) {
#pragma unroll
      for (int r = 0; r < 8; ++r) {
        const int m = row0 + waveM * 64 + mi * 16 + 8 * g + r;
        const int n = col0 + waveN * 32 + ni * 16 + lr;
        if (m < M && n < N) {
          float v = acc[mi][ni][r] * alpha;
          if (FLAGS & F_BIAS) v += bias[n];
          const long idx = oOff + (long)m * ldo + n;
          if (FLAGS & F_RES) v += res[idx];
          if (FLAGS & F_OUTF16) ((_Float16*)out)[idx] = (_Float16)v;
          else                  ((float*)out)[idx]    = v;
        }
      }
    }
  }
}

// ---------------------------------------------------------------------------
// RMSNorm: one 256-thread block per row, f32 in -> f16 out
// ---------------------------------------------------------------------------
__global__ __launch_bounds__(256) void rmsnorm_f16(
    const float* __restrict__ x, const float* __restrict__ scale,
    _Float16* __restrict__ out, int C) {
  const long row = blockIdx.x;
  const float* xr = x + row * (long)C;
  float ss = 0.f;
  for (int i = threadIdx.x; i < C; i += 256) { float v = xr[i]; ss += v * v; }
#pragma unroll
  for (int o = 16; o > 0; o >>= 1) ss += __shfl_xor(ss, o, 32);
  __shared__ float wsum[8];
  const int lane = threadIdx.x & 31, wv = threadIdx.x >> 5;
  if (lane == 0) wsum[wv] = ss;
  __syncthreads();
  if (wv == 0) {
    float t = (lane < 8) ? wsum[lane] : 0.f;
#pragma unroll
    for (int o = 4; o > 0; o >>= 1) t += __shfl_xor(t, o, 32);
    if (lane == 0) wsum[0] = t;
  }
  __syncthreads();
  const float inv = rsqrtf(wsum[0] / (float)C + 1e-5f);
  for (int i = threadIdx.x; i < C; i += 256)
    out[row * (long)C + i] = (_Float16)(xr[i] * scale[i] * inv);
}

// f32 -> f16 cast (weights)
__global__ void cast_f16(const float* __restrict__ in, _Float16* __restrict__ out, long n) {
  const long i = blockIdx.x * 256L + threadIdx.x;
  if (i < n) out[i] = (_Float16)in[i];
}

// RoPE in-place on f16 projections laid out [rows=B*T, C]
__global__ void rope_f16(_Float16* __restrict__ q, const float* __restrict__ cs,
                         const float* __restrict__ sn, long npairs) {
  const long p = blockIdx.x * 256L + threadIdx.x;
  if (p >= npairs) return;
  const int  cp  = (int)(p % (Cq / 2));
  const long row = p / (Cq / 2);
  const int  t   = (int)(row % Tq);
  const int  h   = cp / (HDq / 2);
  const int  d2  = cp % (HDq / 2);
  const long base = row * Cq + h * HDq + 2 * d2;
  const float xr = (float)q[base], xi = (float)q[base + 1];
  const float c = cs[t * (HDq / 2) + d2], s = sn[t * (HDq / 2) + d2];
  q[base]     = (_Float16)(xr * c - xi * s);
  q[base + 1] = (_Float16)(xr * s + xi * c);
}

// cross-attention softmax: rows of length 256, one wave per row
__global__ __launch_bounds__(256) void softmax256(
    const float* __restrict__ sc, _Float16* __restrict__ pr, long nrows) {
  const int lane = threadIdx.x & 31, wv = threadIdx.x >> 5;
  const long row = blockIdx.x * 8L + wv;
  if (row >= nrows) return;
  const float* s = sc + row * 256;
  float v[8], mx = -3.0e38f;
#pragma unroll
  for (int j = 0; j < 8; ++j) { v[j] = s[lane + 32 * j]; mx = fmaxf(mx, v[j]); }
#pragma unroll
  for (int o = 16; o > 0; o >>= 1) mx = fmaxf(mx, __shfl_xor(mx, o, 32));
  float sum = 0.f;
#pragma unroll
  for (int j = 0; j < 8; ++j) { v[j] = __expf(v[j] - mx); sum += v[j]; }
#pragma unroll
  for (int o = 16; o > 0; o >>= 1) sum += __shfl_xor(sum, o, 32);
  const float r = 1.f / sum;
#pragma unroll
  for (int j = 0; j < 8; ++j)
    pr[row * 256 + lane + 32 * j] = (_Float16)(v[j] * r);
}

// self-attention softmax: rows of length T=1024, causal + padding masks
__global__ __launch_bounds__(256) void softmax_self(
    const float* __restrict__ sc, _Float16* __restrict__ pr,
    const unsigned char* __restrict__ amask, const unsigned char* __restrict__ pmask,
    long nrows) {
  const int lane = threadIdx.x & 31, wv = threadIdx.x >> 5;
  const long row = blockIdx.x * 8L + wv;           // row in [0, B*H*T)
  if (row >= nrows) return;
  const int t  = (int)(row % Tq);
  const int b  = (int)(row / ((long)Hq * Tq));
  const float* s = sc + row * (long)Tq;
  float v[32], mx = -3.0e38f;
#pragma unroll
  for (int j = 0; j < 32; ++j) {
    const int col = lane + 32 * j;
    float x = s[col];
    const bool ok = amask[(long)t * Tq + col] && pmask[(long)b * Tq + col];
    x = ok ? x : -1e30f;
    v[j] = x;
    mx = fmaxf(mx, x);
  }
#pragma unroll
  for (int o = 16; o > 0; o >>= 1) mx = fmaxf(mx, __shfl_xor(mx, o, 32));
  float sum = 0.f;
#pragma unroll
  for (int j = 0; j < 32; ++j) { v[j] = __expf(v[j] - mx); sum += v[j]; }
#pragma unroll
  for (int o = 16; o > 0; o >>= 1) sum += __shfl_xor(sum, o, 32);
  const float r = 1.f / sum;
#pragma unroll
  for (int j = 0; j < 32; ++j)
    pr[row * (long)Tq + lane + 32 * j] = (_Float16)(v[j] * r);
}

// h = silu(fc1) * fc2, in place on fc1
__global__ void silu_gate(_Float16* __restrict__ a, const _Float16* __restrict__ b, long n) {
  const long i = blockIdx.x * 256L + threadIdx.x;
  if (i < n) {
    const float x = (float)a[i];
    const float g = x / (1.f + __expf(-x));
    a[i] = (_Float16)(g * (float)b[i]);
  }
}

// ---------------------------------------------------------------------------
extern "C" void kernel_launch(void* const* d_in, const int* in_sizes, int n_in,
                              void* d_out, int out_size, void* d_ws, size_t ws_size,
                              hipStream_t stream) {
  (void)in_sizes; (void)n_in; (void)out_size; (void)ws_size;
  // setup_inputs() dict order
  const float* x     = (const float*)d_in[0];
  const float* y     = (const float*)d_in[1];
  const float* cosT  = (const float*)d_in[2];
  const float* sinT  = (const float*)d_in[3];
  const unsigned char* amask = (const unsigned char*)d_in[4];
  const unsigned char* pmask = (const unsigned char*)d_in[5];
  const float* ln0   = (const float*)d_in[6];
  const float* ln0t  = (const float*)d_in[7];
  const float* ln1   = (const float*)d_in[8];
  const float* ln2   = (const float*)d_in[9];
  const float* wsrc[11] = {
    (const float*)d_in[10], (const float*)d_in[12], (const float*)d_in[14], (const float*)d_in[16],
    (const float*)d_in[18], (const float*)d_in[20], (const float*)d_in[22], (const float*)d_in[24],
    (const float*)d_in[26], (const float*)d_in[27], (const float*)d_in[28] };
  const float* b_ca_q = (const float*)d_in[11];
  const float* b_ca_k = (const float*)d_in[13];
  const float* b_ca_v = (const float*)d_in[15];
  const float* b_ca_o = (const float*)d_in[17];
  const float* b_sa_q = (const float*)d_in[19];
  const float* b_sa_k = (const float*)d_in[21];
  const float* b_sa_v = (const float*)d_in[23];
  const float* b_sa_o = (const float*)d_in[25];

  // ---- workspace carve-out ----
  char* p = (char*)d_ws;
  auto alloc = [&](size_t bytes) -> char* {
    char* r = p;
    p += (bytes + 255) & ~(size_t)255;
    return r;
  };
  const long NCC = (long)Cq * Cq;           // 1M
  const long NFC = (long)FFq * Cq;          // 4M
  const long NXT = (long)Bq * Tq * Cq;      // 4M
  const long NYM = (long)Bq * Mq * Cq;      // 1M
  const long NSS = (long)Bq * Hq * Tq * Tq; // 64M (self scores; cross uses prefix)
  const long NFF = (long)Bq * Tq * FFq;     // 16M

  _Float16* w16[11];
  const long wel[11] = {NCC,NCC,NCC,NCC,NCC,NCC,NCC,NCC,NFC,NFC,NFC};
  for (int i = 0; i < 11; ++i) w16[i] = (_Float16*)alloc((size_t)wel[i] * 2);
  _Float16* xn16   = (_Float16*)alloc((size_t)NXT * 2);
  _Float16* yn16   = (_Float16*)alloc((size_t)NYM * 2);
  _Float16* q16    = (_Float16*)alloc((size_t)NXT * 2);
  _Float16* k16    = (_Float16*)alloc((size_t)NXT * 2);
  _Float16* v16    = (_Float16*)alloc((size_t)NXT * 2);
  _Float16* ctx16  = (_Float16*)alloc((size_t)NXT * 2);
  float*    scores = (float*)   alloc((size_t)NSS * 4);
  _Float16* probs  = (_Float16*)alloc((size_t)NSS * 2);
  float*    xacc   = (float*)   alloc((size_t)NXT * 4);
  _Float16* fc1    = (_Float16*)alloc((size_t)NFF * 2);
  _Float16* fc2    = (_Float16*)alloc((size_t)NFF * 2);

  // ---- 0. weights f32 -> f16 ----
  for (int i = 0; i < 11; ++i)
    cast_f16<<<dim3((unsigned)((wel[i] + 255) / 256)), 256, 0, stream>>>(wsrc[i], w16[i], wel[i]);

  const int rowsX = Bq * Tq;   // 4096
  const int rowsY = Bq * Mq;   // 1024

  // ---- 1. cross-attention ----
  rmsnorm_f16<<<rowsX, 256, 0, stream>>>(x, ln0,  xn16, Cq);
  rmsnorm_f16<<<rowsY, 256, 0, stream>>>(y, ln0t, yn16, Cq);

  gemm_wmma<F_BIAS | F_OUTF16><<<dim3(8, 32, 1), 256, 0, stream>>>(
      xn16, 0, 0, Cq, w16[0], 0, 0, Cq, b_ca_q, nullptr,
      q16, 0, 0, Cq, rowsX, Cq, Cq, 1.f, 1);
  gemm_wmma<F_BIAS | F_OUTF16><<<dim3(8, 8, 1), 256, 0, stream>>>(
      yn16, 0, 0, Cq, w16[1], 0, 0, Cq, b_ca_k, nullptr,
      k16, 0, 0, Cq, rowsY, Cq, Cq, 1.f, 1);
  gemm_wmma<F_BIAS | F_OUTF16><<<dim3(8, 8, 1), 256, 0, stream>>>(
      yn16, 0, 0, Cq, w16[2], 0, 0, Cq, b_ca_v, nullptr,
      v16, 0, 0, Cq, rowsY, Cq, Cq, 1.f, 1);

  // scores[b,h,t,m] = (q . k) / 8 ; batched over B*H via grid.z
  gemm_wmma<0><<<dim3(2, 8, Bq * Hq), 256, 0, stream>>>(
      q16, (long)Tq * Cq, HDq, Cq,
      k16, (long)Mq * Cq, HDq, Cq, nullptr, nullptr,
      scores, (long)Hq * Tq * Mq, (long)Tq * Mq, Mq,
      Tq, Mq, HDq, 0.125f, Hq);
  softmax256<<<(Bq * Hq * Tq) / 8, 256, 0, stream>>>(scores, probs, (long)Bq * Hq * Tq);
  // ctx[b,t,h*HD+d] = probs @ v  (V is [K=m, N=d] -> BTRANS)
  gemm_wmma<F_BTRANS | F_OUTF16><<<dim3(1, 8, Bq * Hq), 256, 0, stream>>>(
      probs, (long)Hq * Tq * Mq, (long)Tq * Mq, Mq,
      v16, (long)Mq * Cq, HDq, Cq, nullptr, nullptr,
      ctx16, (long)Tq * Cq, HDq, Cq,
      Tq, HDq, Mq, 1.f, Hq);
  // xacc = x + ctx @ wo^T + bo
  gemm_wmma<F_BIAS | F_RES><<<dim3(8, 32, 1), 256, 0, stream>>>(
      ctx16, 0, 0, Cq, w16[3], 0, 0, Cq, b_ca_o, x,
      xacc, 0, 0, Cq, rowsX, Cq, Cq, 1.f, 1);

  // ---- 2. self-attention (causal, RoPE) ----
  rmsnorm_f16<<<rowsX, 256, 0, stream>>>(xacc, ln1, xn16, Cq);
  gemm_wmma<F_BIAS | F_OUTF16><<<dim3(8, 32, 1), 256, 0, stream>>>(
      xn16, 0, 0, Cq, w16[4], 0, 0, Cq, b_sa_q, nullptr,
      q16, 0, 0, Cq, rowsX, Cq, Cq, 1.f, 1);
  gemm_wmma<F_BIAS | F_OUTF16><<<dim3(8, 32, 1), 256, 0, stream>>>(
      xn16, 0, 0, Cq, w16[5], 0, 0, Cq, b_sa_k, nullptr,
      k16, 0, 0, Cq, rowsX, Cq, Cq, 1.f, 1);
  gemm_wmma<F_BIAS | F_OUTF16><<<dim3(8, 32, 1), 256, 0, stream>>>(
      xn16, 0, 0, Cq, w16[6], 0, 0, Cq, b_sa_v, nullptr,
      v16, 0, 0, Cq, rowsX, Cq, Cq, 1.f, 1);
  {
    const long npairs = (long)Bq * Tq * Cq / 2;
    rope_f16<<<(unsigned)((npairs + 255) / 256), 256, 0, stream>>>(q16, cosT, sinT, npairs);
    rope_f16<<<(unsigned)((npairs + 255) / 256), 256, 0, stream>>>(k16, cosT, sinT, npairs);
  }
  gemm_wmma<0><<<dim3(8, 8, Bq * Hq), 256, 0, stream>>>(
      q16, (long)Tq * Cq, HDq, Cq,
      k16, (long)Tq * Cq, HDq, Cq, nullptr, nullptr,
      scores, (long)Hq * Tq * Tq, (long)Tq * Tq, Tq,
      Tq, Tq, HDq, 0.125f, Hq);
  softmax_self<<<(Bq * Hq * Tq) / 8, 256, 0, stream>>>(
      scores, probs, amask, pmask, (long)Bq * Hq * Tq);
  gemm_wmma<F_BTRANS | F_OUTF16><<<dim3(1, 8, Bq * Hq), 256, 0, stream>>>(
      probs, (long)Hq * Tq * Tq, (long)Tq * Tq, Tq,
      v16, (long)Tq * Cq, HDq, Cq, nullptr, nullptr,
      ctx16, (long)Tq * Cq, HDq, Cq,
      Tq, HDq, Tq, 1.f, Hq);
  // xacc += ctx @ wo^T + bo   (in-place residual is element-local: safe)
  gemm_wmma<F_BIAS | F_RES><<<dim3(8, 32, 1), 256, 0, stream>>>(
      ctx16, 0, 0, Cq, w16[7], 0, 0, Cq, b_sa_o, xacc,
      xacc, 0, 0, Cq, rowsX, Cq, Cq, 1.f, 1);

  // ---- 3. SwiGLU FFN ----
  rmsnorm_f16<<<rowsX, 256, 0, stream>>>(xacc, ln2, xn16, Cq);
  gemm_wmma<F_OUTF16><<<dim3(32, 32, 1), 256, 0, stream>>>(
      xn16, 0, 0, Cq, w16[8], 0, 0, Cq, nullptr, nullptr,
      fc1, 0, 0, FFq, rowsX, FFq, Cq, 1.f, 1);
  gemm_wmma<F_OUTF16><<<dim3(32, 32, 1), 256, 0, stream>>>(
      xn16, 0, 0, Cq, w16[9], 0, 0, Cq, nullptr, nullptr,
      fc2, 0, 0, FFq, rowsX, FFq, Cq, 1.f, 1);
  silu_gate<<<(unsigned)((NFF + 255) / 256), 256, 0, stream>>>(fc1, fc2, NFF);
  // d_out = xacc + h @ w_proj^T
  gemm_wmma<F_RES><<<dim3(8, 32, 1), 256, 0, stream>>>(
      fc1, 0, 0, FFq, w16[10], 0, 0, FFq, nullptr, xacc,
      d_out, 0, 0, Cq, rowsX, Cq, FFq, 1.f, 1);
}
